// Grapher_36077725286461
// MI455X (gfx1250) — matrix operand order repaired and checked by reading, not compiled, not run on hardware
//
#include <hip/hip_runtime.h>
#include <math.h>

typedef __attribute__((ext_vector_type(16))) _Float16 v16h;
typedef __attribute__((ext_vector_type(8)))  _Float16 v8h;
typedef __attribute__((ext_vector_type(8)))  float    v8f;

#define B_  8
#define C_  96
#define C2_ 192
#define N_  3136
#define NP_ (B_ * N_)   // 25088
#define K_  9

// ---------------------------------------------------------------------------
// WMMA fragment loaders (this code's contiguous k = half*16 + s order, same in A and B)
// A 16x32 f16: lane L, M = L&15, half = L>>4.
//   a[0..7]  = K  (half*16) .. +7      (one 16B chunk)
//   a[8..15] = K (half*16)+8 .. +7     (one 16B chunk)
// B 32x16 f16: lane L, N = L&15; 16 contiguous halfs at K = half*16.
// ---------------------------------------------------------------------------
__device__ __forceinline__ v16h load_afrag(const _Float16* __restrict__ rowks, int half) {
  v16h a;
  ((v8h*)&a)[0] = *(const v8h*)(rowks + half * 16);
  ((v8h*)&a)[1] = *(const v8h*)(rowks + half * 16 + 8);
  return a;
}
__device__ __forceinline__ v16h load_bfrag(const _Float16* __restrict__ p) {
  v16h b;
  ((v8h*)&b)[0] = *(const v8h*)(p);
  ((v8h*)&b)[1] = *(const v8h*)(p + 8);
  return b;
}

// ---------------------------------------------------------------------------
// fp32 -> fp16 convert (weights)
// ---------------------------------------------------------------------------
__global__ void f32_to_f16_kern(const float* __restrict__ in, _Float16* __restrict__ out, int n) {
  int i = blockIdx.x * blockDim.x + threadIdx.x;
  if (i < n) out[i] = (_Float16)in[i];
}

// ---------------------------------------------------------------------------
// x (B,C,N) f32 channel-major -> xt (B,N,C) f16 point-major (LDS transpose)
// ---------------------------------------------------------------------------
__global__ void transpose_in_kern(const float* __restrict__ x, _Float16* __restrict__ xt) {
  __shared__ float t[16][17];
  int b = blockIdx.z, c0 = blockIdx.y * 16, n0 = blockIdx.x * 16;
  int tx = threadIdx.x, ty = threadIdx.y;
  t[ty][tx] = x[((size_t)b * C_ + c0 + ty) * N_ + n0 + tx];
  __syncthreads();
  xt[((size_t)b * N_ + n0 + ty) * C_ + c0 + tx] = (_Float16)t[tx][ty];
}

// ---------------------------------------------------------------------------
// GEMM: out[p][o] = sum_k A[p][k] * W[o][k] + bias[o]
// A point-major f16 (NP x Cin), W row-major f16 (Cout x Cin), out f32 (NP x Cout)
// one wave per 16x16 tile
// ---------------------------------------------------------------------------
__global__ void gemm_wmma_kern(const _Float16* __restrict__ A, const _Float16* __restrict__ W,
                               const float* __restrict__ bias, float* __restrict__ out,
                               int Cin, int Cout) {
  int lane = threadIdx.x;
  int m = lane & 15, half = lane >> 4;
  int pt0 = blockIdx.x * 16;
  int o0  = blockIdx.y * 16;
  const _Float16* arow = A + (size_t)(pt0 + m) * Cin;
  const _Float16* wrow = W + (size_t)(o0 + m) * Cin;
  v8f acc = {0.f, 0.f, 0.f, 0.f, 0.f, 0.f, 0.f, 0.f};
  for (int ks = 0; ks < Cin; ks += 32) {
    v16h a = load_afrag(arow + ks, half);
    v16h b = load_bfrag(wrow + ks + half * 16);
    acc = __builtin_amdgcn_wmma_f32_16x16x32_f16(false, a, false, b, (short)0, acc, false, false);
  }
  float bv = bias[o0 + m];
#pragma unroll
  for (int r = 0; r < 8; ++r) {
    out[(size_t)(pt0 + r + half * 8) * Cout + o0 + m] = acc[r] + bv;
  }
}

// ---------------------------------------------------------------------------
// Deterministic per-channel sum/sumsq over all NP rows (16 channels/block)
// ---------------------------------------------------------------------------
__global__ void bn_stats_kern(const float* __restrict__ pre, float* __restrict__ sum,
                              float* __restrict__ sumsq, int C) {
  __shared__ float ls[64][16];
  __shared__ float lq[64][16];
  int tx = threadIdx.x, ty = threadIdx.y;
  int c = blockIdx.x * 16 + tx;
  float s = 0.f, q = 0.f;
  for (int r = ty; r < NP_; r += 64) {
    float v = pre[(size_t)r * C + c];
    s += v; q += v * v;
  }
  ls[ty][tx] = s; lq[ty][tx] = q;
  __syncthreads();
  for (int off = 32; off > 0; off >>= 1) {
    if (ty < off) { ls[ty][tx] += ls[ty + off][tx]; lq[ty][tx] += lq[ty + off][tx]; }
    __syncthreads();
  }
  if (ty == 0) { sum[c] = ls[0][tx]; sumsq[c] = lq[0][tx]; }
}

__global__ void bn_finalize_kern(const float* __restrict__ sum, const float* __restrict__ sumsq,
                                 const float* __restrict__ g, const float* __restrict__ be,
                                 float* __restrict__ scale, float* __restrict__ shift,
                                 int C, float invCount) {
  int c = blockIdx.x * blockDim.x + threadIdx.x;
  if (c < C) {
    float mu = sum[c] * invCount;
    float var = sumsq[c] * invCount - mu * mu;
    float s = g[c] * rsqrtf(fmaxf(var, 0.f) + 1e-5f);
    scale[c] = s;
    shift[c] = be[c] - mu * s;
  }
}

// ---------------------------------------------------------------------------
// Apply BN to fc1 output, emit h_bn (f16) and L2-normalized xn (f16).
// one wave per point, 3 channels per lane (96 = 3*32)
// ---------------------------------------------------------------------------
__global__ void bn_norm_kern(const float* __restrict__ pre, const float* __restrict__ scale,
                             const float* __restrict__ shift,
                             _Float16* __restrict__ hbn, _Float16* __restrict__ xn) {
  int wave = threadIdx.x >> 5, lane = threadIdx.x & 31;
  int pt = blockIdx.x * 8 + wave;
  const float* row = pre + (size_t)pt * C_;
  float v0 = row[lane]      * scale[lane]      + shift[lane];
  float v1 = row[lane + 32] * scale[lane + 32] + shift[lane + 32];
  float v2 = row[lane + 64] * scale[lane + 64] + shift[lane + 64];
  float s = v0 * v0 + v1 * v1 + v2 * v2;
#pragma unroll
  for (int off = 16; off > 0; off >>= 1) s += __shfl_xor(s, off, 32);
  float inv = 1.f / fmaxf(sqrtf(s), 1e-12f);
  size_t o = (size_t)pt * C_;
  hbn[o + lane]      = (_Float16)v0;
  hbn[o + lane + 32] = (_Float16)v1;
  hbn[o + lane + 64] = (_Float16)v2;
  xn[o + lane]       = (_Float16)(v0 * inv);
  xn[o + lane + 32]  = (_Float16)(v1 * inv);
  xn[o + lane + 64]  = (_Float16)(v2 * inv);
}

// ---------------------------------------------------------------------------
// Fused KNN: S = Xn * Xn^T tile-by-tile via WMMA; per-row top-9 largest dots
// (== 9 smallest L2 distances, self included). One wave = 32 rows of a batch.
// ---------------------------------------------------------------------------
__global__ void knn_topk_kern(const _Float16* __restrict__ xn, int* __restrict__ nn) {
  __shared__ float lds[32 * 16];
  int lane = threadIdx.x;
  int m = lane & 15, half = lane >> 4;
  int b = blockIdx.y;
  int row0 = blockIdx.x * 32;
  const _Float16* Xb = xn + (size_t)b * N_ * C_;

  v16h afr[2][3];
#pragma unroll
  for (int s = 0; s < 2; ++s)
#pragma unroll
    for (int ki = 0; ki < 3; ++ki)
      afr[s][ki] = load_afrag(Xb + (size_t)(row0 + s * 16 + m) * C_ + ki * 32, half);

  float topv[K_];
  int   topi[K_];
#pragma unroll
  for (int j = 0; j < K_; ++j) { topv[j] = -1e30f; topi[j] = 0; }

  for (int ct = 0; ct < N_ / 16; ++ct) {
    int c0 = ct * 16;
    v16h bfr[3];
#pragma unroll
    for (int ki = 0; ki < 3; ++ki)
      bfr[ki] = load_bfrag(Xb + (size_t)(c0 + m) * C_ + ki * 32 + half * 16);

    v8f acc0 = {0.f, 0.f, 0.f, 0.f, 0.f, 0.f, 0.f, 0.f};
    v8f acc1 = {0.f, 0.f, 0.f, 0.f, 0.f, 0.f, 0.f, 0.f};
#pragma unroll
    for (int ki = 0; ki < 3; ++ki)
      acc0 = __builtin_amdgcn_wmma_f32_16x16x32_f16(false, afr[0][ki], false, bfr[ki], (short)0, acc0, false, false);
#pragma unroll
    for (int ki = 0; ki < 3; ++ki)
      acc1 = __builtin_amdgcn_wmma_f32_16x16x32_f16(false, afr[1][ki], false, bfr[ki], (short)0, acc1, false, false);

#pragma unroll
    for (int r = 0; r < 8; ++r) {
      lds[(r + half * 8) * 16 + m]        = acc0[r];
      lds[(16 + r + half * 8) * 16 + m]   = acc1[r];
    }
    __syncthreads();
    // lane owns row (row0+lane); scan 16 candidate columns
#pragma unroll
    for (int j = 0; j < 16; ++j) {
      float v = lds[lane * 16 + j];
      if (v > topv[K_ - 1]) {
        topv[K_ - 1] = v; topi[K_ - 1] = c0 + j;
#pragma unroll
        for (int s = K_ - 1; s > 0; --s) {
          if (topv[s] > topv[s - 1]) {
            float tv = topv[s]; topv[s] = topv[s - 1]; topv[s - 1] = tv;
            int   ti = topi[s]; topi[s] = topi[s - 1]; topi[s - 1] = ti;
          }
        }
      }
    }
    __syncthreads();
  }
  int* dst = nn + ((size_t)b * N_ + row0 + lane) * K_;
#pragma unroll
  for (int j = 0; j < K_; ++j) dst[j] = topi[j];
}

// ---------------------------------------------------------------------------
// Max-relative aggregation + channel interleave:
// y[p][2c] = h[c], y[p][2c+1] = max_j h[nbr_j][c] - h[c]   (packed 2xf16/dword)
// ---------------------------------------------------------------------------
__device__ __forceinline__ unsigned packh(float lo, float hi) {
  union { _Float16 h[2]; unsigned u; } u_;
  u_.h[0] = (_Float16)lo; u_.h[1] = (_Float16)hi;
  return u_.u;
}

__global__ void gather_max_kern(const _Float16* __restrict__ hbn, const int* __restrict__ nn,
                                unsigned* __restrict__ y) {
  int wave = threadIdx.x >> 5, lane = threadIdx.x & 31;
  int p = blockIdx.x * 8 + wave;
  int b = p / N_;
  const _Float16* hrow = hbn + (size_t)p * C_;
  float h0 = (float)hrow[lane], h1 = (float)hrow[lane + 32], h2 = (float)hrow[lane + 64];
  float m0 = -1e30f, m1 = -1e30f, m2 = -1e30f;
  const int* idx = nn + (size_t)p * K_;
#pragma unroll
  for (int j = 0; j < K_; ++j) {
    const _Float16* r = hbn + ((size_t)b * N_ + idx[j]) * C_;
    m0 = fmaxf(m0, (float)r[lane]);
    m1 = fmaxf(m1, (float)r[lane + 32]);
    m2 = fmaxf(m2, (float)r[lane + 64]);
  }
  size_t o = (size_t)p * C_;
  y[o + lane]      = packh(h0, m0 - h0);
  y[o + lane + 32] = packh(h1, m1 - h1);
  y[o + lane + 64] = packh(h2, m2 - h2);
}

// ---------------------------------------------------------------------------
// BN apply (+ exact GELU) -> f16
// ---------------------------------------------------------------------------
__global__ void bn_act_kern(const float* __restrict__ pre, const float* __restrict__ scale,
                            const float* __restrict__ shift, _Float16* __restrict__ out,
                            int total, int C, int gelu) {
  int i = blockIdx.x * blockDim.x + threadIdx.x;
  if (i >= total) return;
  int c = i % C;
  float v = pre[i] * scale[c] + shift[c];
  if (gelu) v = 0.5f * v * (1.f + erff(v * 0.70710678118654752f));
  out[i] = (_Float16)v;
}

// ---------------------------------------------------------------------------
// Final: BN(fc2 pre) transposed back to channel-major + residual
// ---------------------------------------------------------------------------
__global__ void finalize_out_kern(const float* __restrict__ pre, const float* __restrict__ scale,
                                  const float* __restrict__ shift, const float* __restrict__ x,
                                  float* __restrict__ out) {
  __shared__ float t[16][17];
  int b = blockIdx.z, c0 = blockIdx.y * 16, n0 = blockIdx.x * 16;
  int tx = threadIdx.x, ty = threadIdx.y;
  t[ty][tx] = pre[((size_t)b * N_ + n0 + ty) * C_ + c0 + tx];
  __syncthreads();
  int c = c0 + ty, n = n0 + tx;
  size_t oi = ((size_t)b * C_ + c) * N_ + n;
  out[oi] = t[tx][ty] * scale[c] + shift[c] + x[oi];
}

// ---------------------------------------------------------------------------
// Host side
// ---------------------------------------------------------------------------
static inline size_t align256(size_t x) { return (x + 255) & ~(size_t)255; }

extern "C" void kernel_launch(void* const* d_in, const int* in_sizes, int n_in,
                              void* d_out, int out_size, void* d_ws, size_t ws_size,
                              hipStream_t stream) {
  const float* x      = (const float*)d_in[0];
  const float* fc1_w  = (const float*)d_in[1];
  const float* fc1_b  = (const float*)d_in[2];
  const float* fc1_g  = (const float*)d_in[3];
  const float* fc1_be = (const float*)d_in[4];
  const float* g1_w   = (const float*)d_in[5];
  const float* g1_b   = (const float*)d_in[6];
  const float* g1_g   = (const float*)d_in[7];
  const float* g1_be  = (const float*)d_in[8];
  const float* g2_w   = (const float*)d_in[9];
  const float* g2_b   = (const float*)d_in[10];
  const float* g2_g   = (const float*)d_in[11];
  const float* g2_be  = (const float*)d_in[12];
  const float* fc2_w  = (const float*)d_in[13];
  const float* fc2_b  = (const float*)d_in[14];
  const float* fc2_g  = (const float*)d_in[15];
  const float* fc2_be = (const float*)d_in[16];

  char* ws = (char*)d_ws;
  size_t off = 0;
  auto alloc = [&](size_t bytes) -> void* { void* p = ws + off; off = align256(off + bytes); return p; };

  _Float16* wfc1 = (_Float16*)alloc((size_t)C_ * C_ * 2);
  _Float16* wg1  = (_Float16*)alloc((size_t)C2_ * C2_ * 2);
  _Float16* wg2  = (_Float16*)alloc((size_t)C2_ * C2_ * 2);
  _Float16* wfc2 = (_Float16*)alloc((size_t)C_ * C2_ * 2);
  float* sum   = (float*)alloc(C2_ * 4);
  float* sumsq = (float*)alloc(C2_ * 4);
  float* scale = (float*)alloc(C2_ * 4);
  float* shift = (float*)alloc(C2_ * 4);
  _Float16* xt    = (_Float16*)alloc((size_t)NP_ * C_ * 2);
  float*    h_pre = (float*)   alloc((size_t)NP_ * C_ * 4);
  _Float16* hbn   = (_Float16*)alloc((size_t)NP_ * C_ * 2);
  _Float16* xn    = (_Float16*)alloc((size_t)NP_ * C_ * 2);
  int*      nn    = (int*)     alloc((size_t)NP_ * K_ * 4);
  _Float16* y     = (_Float16*)alloc((size_t)NP_ * C2_ * 2);
  float*    pre   = (float*)   alloc((size_t)NP_ * C2_ * 4);   // reused g1/g2/fc2
  _Float16* yact  = (_Float16*)alloc((size_t)NP_ * C2_ * 2);   // reused y1/y2

  const int ptTiles = NP_ / 16;     // 1568
  const float invNP = 1.f / (float)NP_;

  // weights -> f16
  f32_to_f16_kern<<<(C_ * C_ + 255) / 256, 256, 0, stream>>>(fc1_w, wfc1, C_ * C_);
  f32_to_f16_kern<<<(C2_ * C2_ + 255) / 256, 256, 0, stream>>>(g1_w, wg1, C2_ * C2_);
  f32_to_f16_kern<<<(C2_ * C2_ + 255) / 256, 256, 0, stream>>>(g2_w, wg2, C2_ * C2_);
  f32_to_f16_kern<<<(C_ * C2_ + 255) / 256, 256, 0, stream>>>(fc2_w, wfc2, C_ * C2_);

  // x -> point-major f16
  transpose_in_kern<<<dim3(N_ / 16, C_ / 16, B_), dim3(16, 16), 0, stream>>>(x, xt);

  // fc1 + BN
  gemm_wmma_kern<<<dim3(ptTiles, C_ / 16), 32, 0, stream>>>(xt, wfc1, fc1_b, h_pre, C_, C_);
  bn_stats_kern<<<C_ / 16, dim3(16, 64), 0, stream>>>(h_pre, sum, sumsq, C_);
  bn_finalize_kern<<<1, 256, 0, stream>>>(sum, sumsq, fc1_g, fc1_be, scale, shift, C_, invNP);
  bn_norm_kern<<<NP_ / 8, 256, 0, stream>>>(h_pre, scale, shift, hbn, xn);

  // fused KNN (WMMA distance + top-9)
  knn_topk_kern<<<dim3(N_ / 32, B_), 32, 0, stream>>>(xn, nn);

  // max-relative gather + interleave
  gather_max_kern<<<NP_ / 8, 256, 0, stream>>>(hbn, nn, (unsigned*)y);

  // g1 + BN + GELU
  gemm_wmma_kern<<<dim3(ptTiles, C2_ / 16), 32, 0, stream>>>(y, wg1, g1_b, pre, C2_, C2_);
  bn_stats_kern<<<C2_ / 16, dim3(16, 64), 0, stream>>>(pre, sum, sumsq, C2_);
  bn_finalize_kern<<<1, 256, 0, stream>>>(sum, sumsq, g1_g, g1_be, scale, shift, C2_, invNP);
  bn_act_kern<<<((size_t)NP_ * C2_ + 255) / 256, 256, 0, stream>>>(pre, scale, shift, yact,
                                                                   NP_ * C2_, C2_, 1);
  // g2 + BN + GELU
  gemm_wmma_kern<<<dim3(ptTiles, C2_ / 16), 32, 0, stream>>>(yact, wg2, g2_b, pre, C2_, C2_);
  bn_stats_kern<<<C2_ / 16, dim3(16, 64), 0, stream>>>(pre, sum, sumsq, C2_);
  bn_finalize_kern<<<1, 256, 0, stream>>>(sum, sumsq, g2_g, g2_be, scale, shift, C2_, invNP);
  bn_act_kern<<<((size_t)NP_ * C2_ + 255) / 256, 256, 0, stream>>>(pre, scale, shift, yact,
                                                                   NP_ * C2_, C2_, 1);
  // fc2 + BN + residual (transposed write-back)
  gemm_wmma_kern<<<dim3(ptTiles, C_ / 16), 32, 0, stream>>>(yact, wfc2, fc2_b, pre, C2_, C_);
  bn_stats_kern<<<C_ / 16, dim3(16, 64), 0, stream>>>(pre, sum, sumsq, C_);
  bn_finalize_kern<<<1, 256, 0, stream>>>(sum, sumsq, fc2_g, fc2_be, scale, shift, C_, invNP);
  finalize_out_kern<<<dim3(N_ / 16, C_ / 16, B_), dim3(16, 16), 0, stream>>>(pre, scale, shift, x,
                                                                             (float*)d_out);
}
